// Extractor_41979010351339
// MI455X (gfx1250) — compile-verified, matrix operand order checked by reading, not run on hardware
//
#include <hip/hip_runtime.h>
#include <math.h>

// ---------------------------------------------------------------------------
// Problem constants (mirror the reference)
// ---------------------------------------------------------------------------
#define B_   2
#define DIM  1024
#define NH   16
#define NP   4
#define HD   32
#define VD   512          // NH*HD
#define HF   64
#define WF   64
#define HW   (HF*WF)      // 4096
#define NQ   21504
#define HID  256
#define MQ   (B_*NQ)      // 43008  (multiple of 128)
#define MF   (B_*HW)      // 8192

typedef __attribute__((ext_vector_type(16))) _Float16 v16h;
typedef __attribute__((ext_vector_type(8)))  _Float16 v8h;
typedef __attribute__((ext_vector_type(8)))  float    v8f;

// ---------------------------------------------------------------------------
// Pack fp32 weight W[K,N] into fp16 WMMA-B-fragment-major layout:
//   P[((nTile*(K/32) + kBlk)*32 + lane)*16 + e] = W[k, n]
// where n = nTile*16 + (lane&15), half = lane>>4, j = e>>1,
//       k = kBlk*32 + (j>>2)*16 + half*8 + (j&3)*2 + (e&1)
// Each lane's 16-half fragment is contiguous (32 B) -> two b128 loads in GEMM.
// ---------------------------------------------------------------------------
__global__ void k_pack_w(const float* __restrict__ W, _Float16* __restrict__ P,
                         int K, int N) {
  const int t = blockIdx.x * blockDim.x + threadIdx.x;
  if (t >= K * N) return;
  const int e     = t & 15;
  const int lane  = (t >> 4) & 31;
  const int blk   = t >> 9;
  const int kBlks = K >> 5;
  const int nTile = blk / kBlks;
  const int kBlk  = blk % kBlks;
  const int half  = lane >> 4;
  const int n     = nTile * 16 + (lane & 15);
  const int j     = e >> 1;
  const int kb    = ((j >> 2) << 4) + (half << 3) + ((j & 3) << 1) + (e & 1);
  const int k     = kBlk * 32 + kb;
  P[t] = (_Float16)W[(size_t)k * N + n];
}

// ---------------------------------------------------------------------------
// LayerNorm over DIM=1024, one row per block (256 threads), fp16 output
// ---------------------------------------------------------------------------
__global__ void k_ln_f16(const float* __restrict__ x, const float* __restrict__ w,
                         const float* __restrict__ b, _Float16* __restrict__ y) {
  __shared__ float red[256];
  const int tid = threadIdx.x;
  const float* xr = x + (size_t)blockIdx.x * DIM;
  float v[4], s = 0.f, ss = 0.f;
#pragma unroll
  for (int j = 0; j < 4; ++j) {
    v[j] = xr[j * 256 + tid];
    s += v[j];
    ss += v[j] * v[j];
  }
  red[tid] = s; __syncthreads();
  for (int o = 128; o > 0; o >>= 1) { if (tid < o) red[tid] += red[tid + o]; __syncthreads(); }
  const float mean = red[0] * (1.0f / DIM);
  __syncthreads();
  red[tid] = ss; __syncthreads();
  for (int o = 128; o > 0; o >>= 1) { if (tid < o) red[tid] += red[tid + o]; __syncthreads(); }
  const float var  = red[0] * (1.0f / DIM) - mean * mean;
  const float rstd = rsqrtf(var + 1e-6f);
  _Float16* yr = y + (size_t)blockIdx.x * DIM;
#pragma unroll
  for (int j = 0; j < 4; ++j) {
    const int c = j * 256 + tid;
    yr[c] = (_Float16)((v[j] - mean) * rstd * w[c] + b[c]);
  }
}

// ---------------------------------------------------------------------------
// A fragment load: two aligned 16B chunks map 1:1 onto fragment VGPRs.
// Elements 0..7  <-> K = k0 + half*8 + 0..7        (bytes k0*2 + half*16)
// Elements 8..15 <-> K = k0 + 16 + half*8 + 0..7   (bytes k0*2 + 32 + half*16)
// ---------------------------------------------------------------------------
__device__ __forceinline__ v16h load_afrag(const _Float16* ap, int k0, int half) {
  const v8h lo = *(const v8h*)(ap + k0 + half * 8);
  const v8h hi = *(const v8h*)(ap + k0 + 16 + half * 8);
  v16h r;
#pragma unroll
  for (int i = 0; i < 8; ++i) { r[i] = lo[i]; r[8 + i] = hi[i]; }
  return r;
}

// ---------------------------------------------------------------------------
// WMMA GEMM: C[M,N] = A[M,K](f16) * W[K,N](packed f16 fragments) + bias
// One wave32 computes a 16x64 C strip (4 N-tiles); software-pipelined K loop.
// mode 0: store f16   mode 1: store f32   mode 2: store f32 + residual add
// ---------------------------------------------------------------------------
__global__ void k_gemm_wmma(const _Float16* __restrict__ A, const _Float16* __restrict__ P,
                            const float* __restrict__ bias, void* __restrict__ Cout,
                            const float* __restrict__ residual,
                            int M, int N, int K, int mode) {
  const int lane   = threadIdx.x & 31;
  const int wv     = threadIdx.x >> 5;
  const int tileN0 = blockIdx.x * 4;
  const int tileM  = blockIdx.y * 8 + wv;
  if (tileM * 16 >= M) return;            // wave-uniform; EXEC stays all-1s

  const int half  = lane >> 4;
  const int l16   = lane & 15;
  const int kBlks = K >> 5;

  const _Float16* ap = A + (size_t)(tileM * 16 + l16) * K;
  const v16h* Pv = (const v16h*)P;        // 32B-aligned fragment array
  const v16h* p0 = Pv + (size_t)(tileN0 + 0) * kBlks * 32 + lane;
  const v16h* p1 = Pv + (size_t)(tileN0 + 1) * kBlks * 32 + lane;
  const v16h* p2 = Pv + (size_t)(tileN0 + 2) * kBlks * 32 + lane;
  const v16h* p3 = Pv + (size_t)(tileN0 + 3) * kBlks * 32 + lane;

  v8f acc0 = {}, acc1 = {}, acc2 = {}, acc3 = {};

  // software pipeline: preload fragment kb+1 before consuming fragment kb
  v16h ac = load_afrag(ap, 0, half);
  v16h b0 = p0[0], b1 = p1[0], b2 = p2[0], b3 = p3[0];
#pragma unroll 2
  for (int kb = 0; kb < kBlks - 1; ++kb) {
    const size_t nx = (size_t)(kb + 1) * 32;
    const v16h an  = load_afrag(ap, (kb + 1) * 32, half);
    const v16h bn0 = p0[nx];
    const v16h bn1 = p1[nx];
    const v16h bn2 = p2[nx];
    const v16h bn3 = p3[nx];
    acc0 = __builtin_amdgcn_wmma_f32_16x16x32_f16(false, ac, false, b0, (short)0, acc0, false, false);
    acc1 = __builtin_amdgcn_wmma_f32_16x16x32_f16(false, ac, false, b1, (short)0, acc1, false, false);
    acc2 = __builtin_amdgcn_wmma_f32_16x16x32_f16(false, ac, false, b2, (short)0, acc2, false, false);
    acc3 = __builtin_amdgcn_wmma_f32_16x16x32_f16(false, ac, false, b3, (short)0, acc3, false, false);
    ac = an; b0 = bn0; b1 = bn1; b2 = bn2; b3 = bn3;
  }
  acc0 = __builtin_amdgcn_wmma_f32_16x16x32_f16(false, ac, false, b0, (short)0, acc0, false, false);
  acc1 = __builtin_amdgcn_wmma_f32_16x16x32_f16(false, ac, false, b1, (short)0, acc1, false, false);
  acc2 = __builtin_amdgcn_wmma_f32_16x16x32_f16(false, ac, false, b2, (short)0, acc2, false, false);
  acc3 = __builtin_amdgcn_wmma_f32_16x16x32_f16(false, ac, false, b3, (short)0, acc3, false, false);

  const v8f* accs[4] = {&acc0, &acc1, &acc2, &acc3};
#pragma unroll
  for (int t = 0; t < 4; ++t) {
    const int col = (tileN0 + t) * 16 + l16;
    const float bv = bias ? bias[col] : 0.f;
    const v8f a = *accs[t];
#pragma unroll
    for (int r = 0; r < 8; ++r) {
      const int row  = tileM * 16 + r + 8 * half;   // C/D: m = vgpr + 8*(lane>>4)
      const size_t idx = (size_t)row * N + col;
      const float v = a[r] + bv;
      if (mode == 0)       ((_Float16*)Cout)[idx] = (_Float16)v;
      else if (mode == 1)  ((float*)Cout)[idx] = v;
      else                 ((float*)Cout)[idx] = residual[idx] + v;
    }
  }
}

// ---------------------------------------------------------------------------
// Deformable sampling: one wave per (b, q, head); lane = channel (HD == 32)
// ---------------------------------------------------------------------------
__device__ __forceinline__ float tap(const _Float16* vb, int xi, int yi, float w) {
  if (xi < 0 || xi >= WF || yi < 0 || yi >= HF) return 0.f;
  return w * (float)vb[(size_t)(yi * WF + xi) * VD];
}

__global__ void k_sample(const float* __restrict__ refp, const _Float16* __restrict__ value,
                         const float* __restrict__ off, const float* __restrict__ aw,
                         _Float16* __restrict__ outv) {
  const int lane = threadIdx.x & 31;
  const int gw = blockIdx.x * (blockDim.x >> 5) + (threadIdx.x >> 5);
  if (gw >= B_ * NQ * NH) return;
  const int h  = gw % NH;
  const int bq = gw / NH;               // b*NQ + q
  const int b  = bq / NQ;

  const float* op = off + (size_t)bq * (NH * NP * 2) + h * NP * 2;
  const float* ap = aw  + (size_t)bq * (NH * NP)     + h * NP;
  const float rx = refp[(size_t)bq * 2 + 0];
  const float ry = refp[(size_t)bq * 2 + 1];

  // softmax over the NP=4 attention logits
  const float a0 = ap[0], a1 = ap[1], a2 = ap[2], a3 = ap[3];
  const float mx = fmaxf(fmaxf(a0, a1), fmaxf(a2, a3));
  const float e0 = __expf(a0 - mx), e1 = __expf(a1 - mx);
  const float e2 = __expf(a2 - mx), e3 = __expf(a3 - mx);
  const float inv = 1.0f / (e0 + e1 + e2 + e3);
  const float wgt[NP] = {e0 * inv, e1 * inv, e2 * inv, e3 * inv};

  const _Float16* vb = value + (size_t)b * HW * VD + h * HD + lane;
  float acc = 0.f;
#pragma unroll
  for (int p = 0; p < NP; ++p) {
    const float x = (rx + op[p * 2 + 0] * (1.0f / WF)) * WF - 0.5f;
    const float y = (ry + op[p * 2 + 1] * (1.0f / HF)) * HF - 0.5f;
    const float x0f = floorf(x), y0f = floorf(y);
    const float fx = x - x0f, fy = y - y0f;
    const int x0 = (int)x0f, y0 = (int)y0f;
    acc += wgt[p] * (tap(vb, x0,     y0,     (1.f - fx) * (1.f - fy)) +
                     tap(vb, x0 + 1, y0,     fx * (1.f - fy)) +
                     tap(vb, x0,     y0 + 1, (1.f - fx) * fy) +
                     tap(vb, x0 + 1, y0 + 1, fx * fy));
  }
  outv[(size_t)bq * VD + h * HD + lane] = (_Float16)acc;
}

// ---------------------------------------------------------------------------
// Depthwise 3x3 (per pyramid level, SAME/zeros) + bias + exact GELU -> fp16
// ---------------------------------------------------------------------------
__global__ void k_dwconv_gelu(const float* __restrict__ h, const float* __restrict__ dwW,
                              const float* __restrict__ dwb, _Float16* __restrict__ h2) {
  const size_t t = (size_t)blockIdx.x * blockDim.x + threadIdx.x;
  if (t >= (size_t)B_ * NQ * HID) return;
  const int c  = (int)(t % HID);
  const int nq = (int)((t / HID) % NQ);
  const int b  = (int)(t / ((size_t)HID * NQ));

  int Hh, Ww, base;
  if (nq < 16384)      { Hh = 128; Ww = 128; base = 0;     }
  else if (nq < 20480) { Hh = 64;  Ww = 64;  base = 16384; }
  else                 { Hh = 32;  Ww = 32;  base = 20480; }
  const int pos = nq - base;
  const int yy = pos / Ww, xx = pos % Ww;

  const float* hb = h + ((size_t)b * NQ + base) * HID + c;
  float s = dwb[c];
#pragma unroll
  for (int dy = -1; dy <= 1; ++dy)
#pragma unroll
    for (int dx = -1; dx <= 1; ++dx) {
      const int y = yy + dy, x = xx + dx;
      if (y < 0 || y >= Hh || x < 0 || x >= Ww) continue;
      s += hb[(size_t)(y * Ww + x) * HID] * dwW[c * 9 + (dy + 1) * 3 + (dx + 1)];
    }
  h2[t] = (_Float16)(0.5f * s * (1.0f + erff(s * 0.70710678118f)));
}

// ---------------------------------------------------------------------------
// Launch sequence
// ---------------------------------------------------------------------------
extern "C" void kernel_launch(void* const* d_in, const int* in_sizes, int n_in,
                              void* d_out, int out_size, void* d_ws, size_t ws_size,
                              hipStream_t stream) {
  (void)in_sizes; (void)n_in; (void)out_size; (void)ws_size;
  const float* query = (const float*)d_in[0];
  const float* refp  = (const float*)d_in[1];
  const float* feat  = (const float*)d_in[2];
  const float* qn_w  = (const float*)d_in[5],  *qn_b  = (const float*)d_in[6];
  const float* fn_w  = (const float*)d_in[7],  *fn_b  = (const float*)d_in[8];
  const float* fnn_w = (const float*)d_in[9],  *fnn_b = (const float*)d_in[10];
  const float* off_W = (const float*)d_in[11], *off_b = (const float*)d_in[12];
  const float* aw_W  = (const float*)d_in[13], *aw_b  = (const float*)d_in[14];
  const float* vp_W  = (const float*)d_in[15], *vp_b  = (const float*)d_in[16];
  const float* op_W  = (const float*)d_in[17], *op_b  = (const float*)d_in[18];
  const float* fc1_W = (const float*)d_in[19], *fc1_b = (const float*)d_in[20];
  const float* dw_W  = (const float*)d_in[21], *dw_b  = (const float*)d_in[22];
  const float* fc2_W = (const float*)d_in[23], *fc2_b = (const float*)d_in[24];
  float* out = (float*)d_out;

  // workspace sub-allocation (256B aligned)
  char* ws = (char*)d_ws;
  size_t cur = 0;
  auto alloc = [&](size_t bytes) -> void* {
    void* p = ws + cur;
    cur += (bytes + 255) & ~(size_t)255;
    return p;
  };
  _Float16* w_vp  = (_Float16*)alloc((size_t)DIM * VD  * 2);
  _Float16* w_off = (_Float16*)alloc((size_t)DIM * 128 * 2);
  _Float16* w_aw  = (_Float16*)alloc((size_t)DIM * 64  * 2);
  _Float16* w_op  = (_Float16*)alloc((size_t)VD  * DIM * 2);
  _Float16* w_fc1 = (_Float16*)alloc((size_t)DIM * HID * 2);
  _Float16* w_fc2 = (_Float16*)alloc((size_t)HID * DIM * 2);
  _Float16* qln   = (_Float16*)alloc((size_t)MQ * DIM * 2);
  _Float16* fln   = (_Float16*)alloc((size_t)MF * DIM * 2);
  _Float16* val16 = (_Float16*)alloc((size_t)MF * VD  * 2);
  float*    offb  = (float*)   alloc((size_t)MQ * 128 * 4);
  float*    awb   = (float*)   alloc((size_t)MQ * 64  * 4);
  _Float16* outv  = (_Float16*)alloc((size_t)MQ * VD  * 2);
  _Float16* h16   = (_Float16*)alloc((size_t)MQ * DIM * 2);
  float*    hbuf  = (float*)   alloc((size_t)MQ * HID * 4);
  _Float16* h2    = (_Float16*)alloc((size_t)MQ * HID * 2);

  auto pack = [&](const float* s, _Float16* d, int K, int N) {
    k_pack_w<<<dim3((K * N + 255) / 256), dim3(256), 0, stream>>>(s, d, K, N);
  };
  pack(vp_W,  w_vp,  DIM, VD);
  pack(off_W, w_off, DIM, 128);
  pack(aw_W,  w_aw,  DIM, 64);
  pack(op_W,  w_op,  VD,  DIM);
  pack(fc1_W, w_fc1, DIM, HID);
  pack(fc2_W, w_fc2, HID, DIM);

  // LayerNorms -> fp16 activations
  k_ln_f16<<<dim3(MQ), dim3(256), 0, stream>>>(query, qn_w, qn_b, qln);
  k_ln_f16<<<dim3(MF), dim3(256), 0, stream>>>(feat,  fn_w, fn_b, fln);

  // value = LN(feat) @ vp_W + vp_b              (8192 x 512 x 1024) -> f16
  k_gemm_wmma<<<dim3(VD / 64, MF / 128), dim3(256), 0, stream>>>(
      fln, w_vp, vp_b, val16, nullptr, MF, VD, DIM, 0);
  // offsets / attention logits                  (43008 x {128,64} x 1024) -> f32
  k_gemm_wmma<<<dim3(128 / 64, MQ / 128), dim3(256), 0, stream>>>(
      qln, w_off, off_b, offb, nullptr, MQ, 128, DIM, 1);
  k_gemm_wmma<<<dim3(64 / 64, MQ / 128), dim3(256), 0, stream>>>(
      qln, w_aw, aw_b, awb, nullptr, MQ, 64, DIM, 1);

  // deformable bilinear sampling (one wave per (b,q,head))
  k_sample<<<dim3((B_ * NQ * NH) / 8), dim3(256), 0, stream>>>(refp, val16, offb, awb, outv);

  // q = query + sampled @ op_W + op_b           (43008 x 1024 x 512) -> d_out
  k_gemm_wmma<<<dim3(DIM / 64, MQ / 128), dim3(256), 0, stream>>>(
      outv, w_op, op_b, out, query, MQ, DIM, VD, 2);

  // ConvFFN: h = LN(q) @ fc1 ; dwconv+GELU ; out = q + h2 @ fc2
  k_ln_f16<<<dim3(MQ), dim3(256), 0, stream>>>(out, fnn_w, fnn_b, h16);
  k_gemm_wmma<<<dim3(HID / 64, MQ / 128), dim3(256), 0, stream>>>(
      h16, w_fc1, fc1_b, hbuf, nullptr, MQ, HID, DIM, 1);
  k_dwconv_gelu<<<dim3((MQ * HID) / 256), dim3(256), 0, stream>>>(hbuf, dw_W, dw_b, h2);
  k_gemm_wmma<<<dim3(DIM / 64, MQ / 128), dim3(256), 0, stream>>>(
      h2, w_fc2, fc2_b, out, out, MQ, DIM, HID, 2);
}
